// DbrxFFN_65816078844560
// MI455X (gfx1250) — matrix operand order repaired and checked by reading, not compiled, not run on hardware
//
#include <hip/hip_runtime.h>
#include <hip/hip_bf16.h>
#include <cstdint>

// ---------------- problem constants ----------------
#define TOK    4096      // B*S
#define DMODEL 1024
#define NEXP   8
#define FFN    2048
#define TOPK   2

typedef __attribute__((ext_vector_type(16))) __bf16 v16bf;
typedef __attribute__((ext_vector_type(8)))  float  v8f;

// ---------------- workspace layout (bytes) ----------------
static constexpr size_t WS_META   = 0;           // 64 ints; meta[0]=total 32-row tiles
static constexpr size_t WS_TILE_E = 256;         // 512 ints
static constexpr size_t WS_TILE_S = 2304;        // 512 ints
static constexpr size_t WS_COUNTS = 4352;        // 8 ints
static constexpr size_t WS_TOPKE  = 4608;        // TOK*2 ints
static constexpr size_t WS_TOPKG  = 37376;       // TOK*2 floats
static constexpr size_t WS_ELIST  = 70144;       // NEXP*TOK ints
static constexpr size_t WS_ASSIGN = 201216;      // 8448 ints
static constexpr size_t WS_GATE   = 235008;      // 8448 floats
static constexpr size_t WS_HB     = 268800;      // hidden bf16: 8 MB
static constexpr size_t WS_W1B    = 8657408;     // w1 bf16: 33.55 MB
static constexpr size_t WS_V1B    = 42211840;    // v1 bf16: 33.55 MB
static constexpr size_t WS_W2T    = 75766272;    // w2^T bf16 [E][D][F]: 33.55 MB
static constexpr size_t WS_H      = 109320704;   // 8448*FFN bf16: 34.6 MB
static constexpr size_t WS_YPART  = 143923712;   // TOK*TOPK*DMODEL f32: 33.55 MB

#define MAX_TILES 264    // 8192/32 + 8 experts' padding

// ---------------- WMMA helpers ----------------
__device__ inline v8f wmma_bf16(v16bf a, v16bf b, v8f c) {
  return __builtin_amdgcn_wmma_f32_16x16x32_bf16(false, a, false, b,
                                                 (short)0, c, false, false);
}

// A-fragment from LDS bf16 tile (row-major, rowStride in halves).
// lane L: row = L&15, halves [k0+kb .. +7] and [k0+kb+16 .. +23], kb = L<16?0:8
__device__ inline v16bf load_a_frag(const __bf16* base, int rowStride, int k0) {
  const int lane = threadIdx.x & 31;
  const __bf16* p = base + (size_t)(lane & 15) * rowStride + k0 + ((lane < 16) ? 0 : 8);
  union { uint4 u[2]; v16bf v; } r;
  r.u[0] = *(const uint4*)(p);        // ds_load_b128
  r.u[1] = *(const uint4*)(p + 16);   // ds_load_b128
  return r.v;
}

// B-fragment: bf16 [kb..kb+7] and [kb+16..kb+23] for this lane (column n = lane&15, kb = 8 by lane half)
__device__ inline v16bf load_b_frag(const __bf16* p) {
  union { uint4 u[2]; v16bf v; } r;
  r.u[0] = *(const uint4*)(p);        // global_load_b128
  r.u[1] = *(const uint4*)(p + 16);   // global_load_b128
  return r.v;
}

// ---------------- 0a) elementwise f32 -> bf16 (one float4 per thread) ----------------
__global__ __launch_bounds__(256) void cvt_bf16_kernel(const float* __restrict__ src,
                                                       __bf16* __restrict__ dst) {
  const size_t i = (size_t)blockIdx.x * 256 + threadIdx.x;
  const float4 x = ((const float4*)src)[i];
  union { uint2 u; __bf16 h[4]; } pk;
  pk.h[0] = (__bf16)x.x; pk.h[1] = (__bf16)x.y;
  pk.h[2] = (__bf16)x.z; pk.h[3] = (__bf16)x.w;
  ((uint2*)dst)[i] = pk.u;
}

// ---------------- 0b) w2 [E][F][D] f32 -> w2t [E][D][F] bf16 (LDS 64x64 transpose) ----
__global__ __launch_bounds__(256) void w2_transpose_kernel(const float* __restrict__ w2,
                                                           __bf16* __restrict__ w2t) {
  __shared__ __bf16 s[64][72];                       // padded to dodge bank conflicts
  const int e  = blockIdx.y;
  const int tf = (blockIdx.x & 31) * 64;             // F/64 = 32 tiles
  const int td = (blockIdx.x >> 5) * 64;             // D/64 = 16 tiles
  const size_t ibase = (size_t)e * FFN * DMODEL;
  const int tid = threadIdx.x;
#pragma unroll
  for (int j = 0; j < 4; ++j) {
    const int lin = tid + j * 256;                   // float4 units: 16 per f-row
    const int r = lin >> 4, c4 = (lin & 15) * 4;
    const float4 x = *(const float4*)(w2 + ibase + (size_t)(tf + r) * DMODEL + td + c4);
    s[c4 + 0][r] = (__bf16)x.x; s[c4 + 1][r] = (__bf16)x.y;
    s[c4 + 2][r] = (__bf16)x.z; s[c4 + 3][r] = (__bf16)x.w;
  }
  __syncthreads();
  const size_t obase = (size_t)e * DMODEL * FFN;
#pragma unroll
  for (int j = 0; j < 2; ++j) {
    const int lin = tid + j * 256;                   // uint4 units: 8 per d-row
    const int r = lin >> 3, c8 = (lin & 7) * 8;
    union { uint4 u; __bf16 h[8]; } pk;
#pragma unroll
    for (int i = 0; i < 8; ++i) pk.h[i] = s[r][c8 + i];
    *(uint4*)(w2t + obase + (size_t)(td + r) * FFN + tf + c8) = pk.u;
  }
}

// ---------------- 1) router: logits -> softmax -> top2 -> gates ----------------
__global__ __launch_bounds__(256) void router_kernel(
    const float* __restrict__ hidden, const float* __restrict__ rk,
    float* __restrict__ weights_out, int* __restrict__ topk_e,
    float* __restrict__ topk_g) {
  const int wave = threadIdx.x >> 5, lane = threadIdx.x & 31;
  const int t = blockIdx.x * 8 + wave;               // 512 blocks * 8 waves
  float acc[NEXP];
#pragma unroll
  for (int e = 0; e < NEXP; ++e) acc[e] = 0.f;
  const float* hp = hidden + (size_t)t * DMODEL;
#pragma unroll 4
  for (int i = 0; i < DMODEL / 32; ++i) {
    const int d = lane + i * 32;
    const float x = hp[d];
    const float4 r0 = *(const float4*)(rk + (size_t)d * NEXP);
    const float4 r1 = *(const float4*)(rk + (size_t)d * NEXP + 4);
    acc[0] += x * r0.x; acc[1] += x * r0.y; acc[2] += x * r0.z; acc[3] += x * r0.w;
    acc[4] += x * r1.x; acc[5] += x * r1.y; acc[6] += x * r1.z; acc[7] += x * r1.w;
  }
#pragma unroll
  for (int off = 16; off > 0; off >>= 1)
#pragma unroll
    for (int e = 0; e < NEXP; ++e) acc[e] += __shfl_xor(acc[e], off, 32);
  if (lane == 0) {
    float m = acc[0];
#pragma unroll
    for (int e = 1; e < NEXP; ++e) m = fmaxf(m, acc[e]);
    float s = 0.f, w[NEXP];
#pragma unroll
    for (int e = 0; e < NEXP; ++e) { w[e] = __expf(acc[e] - m); s += w[e]; }
    const float inv = 1.f / s;
#pragma unroll
    for (int e = 0; e < NEXP; ++e) { w[e] *= inv; weights_out[(size_t)t * NEXP + e] = w[e]; }
    int i0 = 0;
#pragma unroll
    for (int e = 1; e < NEXP; ++e) if (w[e] > w[i0]) i0 = e;   // ties -> lower index
    int i1 = (i0 == 0) ? 1 : 0;
#pragma unroll
    for (int e = 0; e < NEXP; ++e) if (e != i0 && w[e] > w[i1]) i1 = e;
    const float norm = 1.f / (w[i0] + w[i1]);
    topk_e[2 * t + 0] = i0; topk_g[2 * t + 0] = w[i0] * norm;
    topk_e[2 * t + 1] = i1; topk_g[2 * t + 1] = w[i1] * norm;
  }
}

// ---------------- 2) per-expert deterministic compaction ----------------
__global__ __launch_bounds__(256) void build_lists(
    const int* __restrict__ topk_e, int* __restrict__ elist, int* __restrict__ counts) {
  const int e = blockIdx.x, tid = threadIdx.x;
  __shared__ int scan[256];
  const int t0 = tid * (TOK / 256);
  int cnt = 0;
#pragma unroll
  for (int i = 0; i < TOK / 256; ++i) {
    const int t = t0 + i;
    cnt += (topk_e[2 * t] == e) + (topk_e[2 * t + 1] == e);
  }
  scan[tid] = cnt;
  __syncthreads();
  for (int off = 1; off < 256; off <<= 1) {
    int v = (tid >= off) ? scan[tid - off] : 0;
    __syncthreads();
    scan[tid] += v;
    __syncthreads();
  }
  int pos = scan[tid] - cnt;
  int* lst = elist + (size_t)e * TOK;
#pragma unroll
  for (int i = 0; i < TOK / 256; ++i) {
    const int t = t0 + i;
    if (topk_e[2 * t]     == e) lst[pos++] = 2 * t;
    if (topk_e[2 * t + 1] == e) lst[pos++] = 2 * t + 1;
  }
  if (tid == 255) counts[e] = scan[255];
}

// ---------------- 3) tile tables + flat assignment/gate lists ----------------
__global__ __launch_bounds__(256) void build_tables(
    const int* __restrict__ counts, const int* __restrict__ elist,
    const float* __restrict__ topk_g, int* __restrict__ meta,
    int* __restrict__ tile_e, int* __restrict__ tile_s,
    int* __restrict__ assign, float* __restrict__ gate) {
  __shared__ int sc[NEXP], sb[NEXP];
  if (threadIdx.x == 0) {
    int base = 0, nt = 0;
    for (int e = 0; e < NEXP; ++e) {
      const int c = counts[e];
      sc[e] = c; sb[e] = base;
      const int pc = (c + 31) & ~31;
      for (int i = 0; i < pc; i += 32) { tile_e[nt] = e; tile_s[nt] = base + i; ++nt; }
      base += pc;
    }
    meta[0] = nt; meta[1] = base;
  }
  __syncthreads();
  for (int e = 0; e < NEXP; ++e) {
    const int c = sc[e], pc = (c + 31) & ~31, b = sb[e];
    for (int i = threadIdx.x; i < pc; i += 256) {
      if (i < c) {
        const int a = elist[(size_t)e * TOK + i];
        assign[b + i] = a; gate[b + i] = topk_g[a];
      } else {
        assign[b + i] = -1; gate[b + i] = 0.f;
      }
    }
  }
}

// ---------------- 4) h = gate * silu(X*W1^T) * (X*V1^T)  [WMMA bf16] ----------------
// grid: (MAX_TILES, FFN/256), block 256 (8 waves). dyn LDS = 32*1024*2 + 256 B.
__global__ __launch_bounds__(256) void glu_h_kernel(
    const __bf16* __restrict__ hb, const __bf16* __restrict__ w1b,
    const __bf16* __restrict__ v1b, const int* __restrict__ meta,
    const int* __restrict__ tile_e, const int* __restrict__ tile_s,
    const int* __restrict__ assign, const float* __restrict__ gate,
    __bf16* __restrict__ h_ws) {
  extern __shared__ char smem[];
  __bf16* A       = (__bf16*)smem;                       // [32][1024] bf16
  float*  gate_sm = (float*)(smem + 32 * 1024 * 2);      // [32]
  int*    asn_sm  = (int*)(gate_sm + 32);                // [32]

  if (blockIdx.x >= meta[0]) return;
  const int e = tile_e[blockIdx.x];
  const int slot0 = tile_s[blockIdx.x];
  const int f0 = blockIdx.y * 256;
  const int tid = threadIdx.x;

  if (tid < 32) { asn_sm[tid] = assign[slot0 + tid]; gate_sm[tid] = gate[slot0 + tid]; }
  __syncthreads();
  // stage 32 gathered bf16 token rows (zeros for padding rows): 128 uint4/row
#pragma unroll 4
  for (int j = 0; j < 16; ++j) {
    const int lin = tid + j * 256;
    const int r = lin >> 7, c = lin & 127;
    uint4 v = make_uint4(0u, 0u, 0u, 0u);
    const int a = asn_sm[r];
    if (a >= 0) v = ((const uint4*)(hb + (size_t)(a >> 1) * DMODEL))[c];
    ((uint4*)(A + (size_t)r * DMODEL))[c] = v;
  }
  __syncthreads();

  const int wave = tid >> 5, lane = tid & 31;
  const int n0 = wave * 2;
  const int col = lane & 15;
  const int koff = (lane < 16) ? 0 : 8;
  v8f acc1[2][2], acc2[2][2];
  const v8f vzero = {0.f, 0.f, 0.f, 0.f, 0.f, 0.f, 0.f, 0.f};
#pragma unroll
  for (int nt = 0; nt < 2; ++nt)
#pragma unroll
    for (int mt = 0; mt < 2; ++mt) { acc1[nt][mt] = vzero; acc2[nt][mt] = vzero; }

  const size_t eFD = (size_t)e * FFN * DMODEL;
  const __bf16* pw0 = w1b + eFD + (size_t)(f0 + (n0 + 0) * 16 + col) * DMODEL + koff;
  const __bf16* pw1 = w1b + eFD + (size_t)(f0 + (n0 + 1) * 16 + col) * DMODEL + koff;
  const __bf16* pv0 = v1b + eFD + (size_t)(f0 + (n0 + 0) * 16 + col) * DMODEL + koff;
  const __bf16* pv1 = v1b + eFD + (size_t)(f0 + (n0 + 1) * 16 + col) * DMODEL + koff;
  for (int k0 = 0; k0 < DMODEL; k0 += 32) {
    const v16bf a0 = load_a_frag(A, DMODEL, k0);
    const v16bf a1 = load_a_frag(A + 16 * DMODEL, DMODEL, k0);
    const v16bf bw0 = load_b_frag(pw0 + k0);
    const v16bf bw1 = load_b_frag(pw1 + k0);
    const v16bf bv0 = load_b_frag(pv0 + k0);
    const v16bf bv1 = load_b_frag(pv1 + k0);
    acc1[0][0] = wmma_bf16(a0, bw0, acc1[0][0]);
    acc1[0][1] = wmma_bf16(a1, bw0, acc1[0][1]);
    acc1[1][0] = wmma_bf16(a0, bw1, acc1[1][0]);
    acc1[1][1] = wmma_bf16(a1, bw1, acc1[1][1]);
    acc2[0][0] = wmma_bf16(a0, bv0, acc2[0][0]);
    acc2[0][1] = wmma_bf16(a1, bv0, acc2[0][1]);
    acc2[1][0] = wmma_bf16(a0, bv1, acc2[1][0]);
    acc2[1][1] = wmma_bf16(a1, bv1, acc2[1][1]);
  }
  // epilogue: silu(x1)*x2*gate -> bf16 h
  const int rbase = (lane < 16) ? 0 : 8;
#pragma unroll
  for (int nt = 0; nt < 2; ++nt) {
    const int fcol = f0 + (n0 + nt) * 16 + col;
#pragma unroll
    for (int mt = 0; mt < 2; ++mt)
#pragma unroll
      for (int r = 0; r < 8; ++r) {
        const int m = mt * 16 + rbase + r;
        if (asn_sm[m] >= 0) {
          const float x1 = acc1[nt][mt][r], x2 = acc2[nt][mt][r];
          const float hv = (x1 / (1.f + __expf(-x1))) * x2 * gate_sm[m];
          h_ws[(size_t)(slot0 + m) * FFN + fcol] = (__bf16)hv;
        }
      }
  }
}

// ---------------- 5) Y = h * W2 (w2t bf16 [E][D][F])  [WMMA bf16] ----------------
// grid: (MAX_TILES, DMODEL/256), block 256. dyn LDS = 32*1024*2 + 128 B (two F halves).
__global__ __launch_bounds__(256) void ffn_out_kernel(
    const __bf16* __restrict__ h_ws, const __bf16* __restrict__ w2t,
    const int* __restrict__ meta, const int* __restrict__ tile_e,
    const int* __restrict__ tile_s, const int* __restrict__ assign,
    float* __restrict__ ypart) {
  extern __shared__ char smem[];
  __bf16* Hs = (__bf16*)smem;                            // [32][1024] bf16 (one F half)
  int* asn_sm = (int*)(smem + 32 * 1024 * 2);            // [32]

  if (blockIdx.x >= meta[0]) return;
  const int e = tile_e[blockIdx.x];
  const int slot0 = tile_s[blockIdx.x];
  const int d0 = blockIdx.y * 256;
  const int tid = threadIdx.x;
  if (tid < 32) asn_sm[tid] = assign[slot0 + tid];
  __syncthreads();

  const int wave = tid >> 5, lane = tid & 31;
  const int n0 = wave * 2;
  const int col = lane & 15;
  const int koff = (lane < 16) ? 0 : 8;
  v8f acc[2][2];
  const v8f vzero = {0.f, 0.f, 0.f, 0.f, 0.f, 0.f, 0.f, 0.f};
#pragma unroll
  for (int nt = 0; nt < 2; ++nt)
#pragma unroll
    for (int mt = 0; mt < 2; ++mt) acc[nt][mt] = vzero;

  const size_t eDF = (size_t)e * DMODEL * FFN;
  const __bf16* pb0 = w2t + eDF + (size_t)(d0 + (n0 + 0) * 16 + col) * FFN + koff;
  const __bf16* pb1 = w2t + eDF + (size_t)(d0 + (n0 + 1) * 16 + col) * FFN + koff;
  for (int half = 0; half < 2; ++half) {
    __syncthreads();
#pragma unroll 4
    for (int j = 0; j < 16; ++j) {
      const int lin = tid + j * 256;                     // uint4 units: 128 per row
      const int r = lin >> 7, c = lin & 127;
      uint4 val = make_uint4(0u, 0u, 0u, 0u);
      if (asn_sm[r] >= 0)
        val = ((const uint4*)(h_ws + (size_t)(slot0 + r) * FFN + half * 1024))[c];
      ((uint4*)(Hs + (size_t)r * 1024))[c] = val;
    }
    __syncthreads();
    const int kg0 = half * 1024;
    for (int k0 = 0; k0 < 1024; k0 += 32) {
      const v16bf a0 = load_a_frag(Hs, 1024, k0);
      const v16bf a1 = load_a_frag(Hs + 16 * 1024, 1024, k0);
      const v16bf b0 = load_b_frag(pb0 + kg0 + k0);
      const v16bf b1 = load_b_frag(pb1 + kg0 + k0);
      acc[0][0] = wmma_bf16(a0, b0, acc[0][0]);
      acc[0][1] = wmma_bf16(a1, b0, acc[0][1]);
      acc[1][0] = wmma_bf16(a0, b1, acc[1][0]);
      acc[1][1] = wmma_bf16(a1, b1, acc[1][1]);
    }
  }
  const int rbase = (lane < 16) ? 0 : 8;
#pragma unroll
  for (int nt = 0; nt < 2; ++nt) {
    const int d = d0 + (n0 + nt) * 16 + col;
#pragma unroll
    for (int mt = 0; mt < 2; ++mt)
#pragma unroll
      for (int r = 0; r < 8; ++r) {
        const int m = mt * 16 + rbase + r;
        const int a = asn_sm[m];
        if (a >= 0) ypart[(size_t)a * DMODEL + d] = acc[nt][mt][r];
      }
  }
}

// ---------------- 6) out[t] = y[t,k=0] + y[t,k=1]  (fixed order) ----------------
__global__ __launch_bounds__(256) void combine_kernel(
    const float* __restrict__ ypart, float* __restrict__ out) {
  const size_t i = (size_t)blockIdx.x * 256 + threadIdx.x;   // float4 units
  const size_t t = i >> 8, c = i & 255;                      // 256 float4 per row
  const float4 a = ((const float4*)ypart)[(t * 2)     * 256 + c];
  const float4 b = ((const float4*)ypart)[(t * 2 + 1) * 256 + c];
  ((float4*)out)[i] = make_float4(a.x + b.x, a.y + b.y, a.z + b.z, a.w + b.w);
}

// ---------------- launcher ----------------
extern "C" void kernel_launch(void* const* d_in, const int* in_sizes, int n_in,
                              void* d_out, int out_size, void* d_ws, size_t ws_size,
                              hipStream_t stream) {
  const float* hidden = (const float*)d_in[0];
  const float* rk     = (const float*)d_in[1];
  const float* w1     = (const float*)d_in[2];
  const float* v1     = (const float*)d_in[3];
  const float* w2     = (const float*)d_in[4];
  float* out     = (float*)d_out;
  float* wts_out = out + (size_t)TOK * DMODEL;   // second output: softmax weights

  char* ws = (char*)d_ws;
  int*    meta   = (int*)(ws + WS_META);
  int*    tile_e = (int*)(ws + WS_TILE_E);
  int*    tile_s = (int*)(ws + WS_TILE_S);
  int*    counts = (int*)(ws + WS_COUNTS);
  int*    topk_e = (int*)(ws + WS_TOPKE);
  float*  topk_g = (float*)(ws + WS_TOPKG);
  int*    elist  = (int*)(ws + WS_ELIST);
  int*    assign = (int*)(ws + WS_ASSIGN);
  float*  gate   = (float*)(ws + WS_GATE);
  __bf16* hb     = (__bf16*)(ws + WS_HB);
  __bf16* w1b    = (__bf16*)(ws + WS_W1B);
  __bf16* v1b    = (__bf16*)(ws + WS_V1B);
  __bf16* w2t    = (__bf16*)(ws + WS_W2T);
  __bf16* h_ws   = (__bf16*)(ws + WS_H);
  float*  ypart  = (float*)(ws + WS_YPART);

  // one-time-per-launch precision/layout conversion (mem-bound, ~13 us)
  cvt_bf16_kernel<<<(TOK * DMODEL / 4) / 256, 256, 0, stream>>>(hidden, hb);
  cvt_bf16_kernel<<<(NEXP * FFN * DMODEL / 4) / 256, 256, 0, stream>>>(w1, w1b);
  cvt_bf16_kernel<<<(NEXP * FFN * DMODEL / 4) / 256, 256, 0, stream>>>(v1, v1b);
  w2_transpose_kernel<<<dim3((FFN / 64) * (DMODEL / 64), NEXP), 256, 0, stream>>>(w2, w2t);

  router_kernel<<<TOK / 8, 256, 0, stream>>>(hidden, rk, wts_out, topk_e, topk_g);
  build_lists<<<NEXP, 256, 0, stream>>>(topk_e, elist, counts);
  build_tables<<<1, 256, 0, stream>>>(counts, elist, topk_g, meta,
                                      tile_e, tile_s, assign, gate);

  const size_t smemA = 32 * 1024 * 2 + 256;
  glu_h_kernel<<<dim3(MAX_TILES, FFN / 256), 256, smemA, stream>>>(
      hb, w1b, v1b, meta, tile_e, tile_s, assign, gate, h_ws);
  const size_t smemB = 32 * 1024 * 2 + 128;
  ffn_out_kernel<<<dim3(MAX_TILES, DMODEL / 256), 256, smemB, stream>>>(
      h_ws, w2t, meta, tile_e, tile_s, assign, ypart);
  combine_kernel<<<(TOK * DMODEL / 4) / 256, 256, 0, stream>>>(ypart, out);
}
